// ModelLayer_10986526343795
// MI455X (gfx1250) — matrix operand order repaired and checked by reading, not compiled, not run on hardware
//
#include <hip/hip_runtime.h>
#include <hip/hip_bf16.h>

#define HH  128
#define HH2 256
#define HH3 384
#define NTG 4   // n-tiles (16 cols each) per wave => 64-column strip

typedef __attribute__((ext_vector_type(16))) _Float16 v16h;
typedef __attribute__((ext_vector_type(8)))  float    v8f;

// ---------------------------------------------------------------------------
// Weight pre-pack: W[K,Nc] fp32 row-major -> f16 fragment-native layout.
// Bp[((kt*ntiles + nt)*32 + lane)*16 + e] = W[(kt*32 + 16*(lane>>4) + e)*Nc
//                                             + nt*16 + (lane&15)]
// Each lane's 16 halfs are 32 contiguous bytes -> b128-pair loads in the GEMM.
// ---------------------------------------------------------------------------
__global__ void pack_w(const float* __restrict__ W, _Float16* __restrict__ Bp,
                       int K, int Nc)
{
  long t = (long)blockIdx.x * blockDim.x + threadIdx.x;
  if (t >= (long)K * Nc) return;
  int e    = (int)(t & 15);
  int lane = (int)((t >> 4) & 31);
  long tile = t >> 9;                       // kt*ntiles + nt
  int ntiles = Nc >> 4;
  int kt = (int)(tile / ntiles), nt = (int)(tile % ntiles);
  int half = lane >> 4, l16 = lane & 15;
  int k = kt * 32 + 16 * half + e;
  Bp[t] = (_Float16)W[(long)k * Nc + nt * 16 + l16];
}

// ---------------------------------------------------------------------------
// WMMA GEMM: C[M,Nc] = A[M,K] @ W[K,Nc]  (+= existing C when acc != 0)
// M%16==0, K%32==0, Nc%64==0 (true for all GEMMs in this model).
// One 16x64 output strip per wave. Per k-step: one A fragment, 4 packed B
// fragments prefetched into registers (constant-offset b128 loads off a single
// lane pointer), then 4 back-to-back v_wmma_f32_16x16x32_f16.
// ---------------------------------------------------------------------------
__global__ __launch_bounds__(256) void gemm_wmma(
    const float* __restrict__ A, const _Float16* __restrict__ Bp,
    float* __restrict__ C, int M, int K, int Nc, int acc)
{
  const int lane = threadIdx.x & 31;
  const int wave = threadIdx.x >> 5;
  const int ngroups = Nc >> 6;              // 64-col strips
  const int ntiles  = Nc >> 4;
  long gw = (long)blockIdx.x * 8 + wave;
  long tiles = (long)(M >> 4) * (long)ngroups;
  if (gw >= tiles) return;
  const int mt = (int)(gw / ngroups);
  const int ng = (int)(gw % ngroups);
  const int row0 = mt << 4, col0 = ng << 6;
  const int half = lane >> 4, l16 = lane & 15;

  // all row*ld products fit in 32 bits for this model
  float* Crow = C + (size_t)((row0 + 8 * half) * Nc + col0 + l16);
  v8f c[NTG] = {};
  if (acc) {
#pragma unroll
    for (int j = 0; j < NTG; ++j)
#pragma unroll
      for (int v = 0; v < 8; ++v)
        c[j][v] = Crow[v * Nc + j * 16 - ((v < 8) ? 0 : 0)];
  }
  // NOTE: C layout per 16x16 f32 D: VGPR v -> row (row0 + v + 8*half).
  // Crow above starts at row (row0 + 8*half); element v adds v*Nc.

  const float* Abase = A + (size_t)((row0 + l16) * K) + 16 * half;
  const _Float16* bp = Bp + ((size_t)(ng * NTG) * 32 + lane) * 16;
  const int bstride = ntiles * 512;         // halfs advanced per k-step

  for (int kt = 0; kt < (K >> 5); ++kt) {
    // A fragment (16x32 f16): element e -> k = k0 + 16*half + e
    const float4* ap = (const float4*)(Abase + (kt << 5));
    float4 a0 = ap[0], a1 = ap[1];          // k0+16h .. k0+16h+7
    float4 a2 = ap[2], a3 = ap[3];          // k0+16h+8 .. k0+16h+15
    v16h a;
    a[0]=(_Float16)a0.x; a[1]=(_Float16)a0.y; a[2]=(_Float16)a0.z; a[3]=(_Float16)a0.w;
    a[4]=(_Float16)a1.x; a[5]=(_Float16)a1.y; a[6]=(_Float16)a1.z; a[7]=(_Float16)a1.w;
    a[8]=(_Float16)a2.x; a[9]=(_Float16)a2.y; a[10]=(_Float16)a2.z; a[11]=(_Float16)a2.w;
    a[12]=(_Float16)a3.x; a[13]=(_Float16)a3.y; a[14]=(_Float16)a3.z; a[15]=(_Float16)a3.w;
    // prefetch all 4 B fragments (constant offsets j*512 halfs = j*1024 bytes)
    v16h bf[NTG];
#pragma unroll
    for (int j = 0; j < NTG; ++j)
      bf[j] = *(const v16h*)(bp + j * 512);
#pragma unroll
    for (int j = 0; j < NTG; ++j)
      c[j] = __builtin_amdgcn_wmma_f32_16x16x32_f16(false, a, false, bf[j],
                                                    (short)0, c[j], false, false);
    bp += bstride;
  }
#pragma unroll
  for (int j = 0; j < NTG; ++j)
#pragma unroll
    for (int v = 0; v < 8; ++v)
      Crow[v * Nc + j * 16] = c[j][v];
}

// ---------------------------------------------------------------------------
// BatchNorm (training mode, biased stats) helpers
// ---------------------------------------------------------------------------
__global__ void bn_stats(const float* __restrict__ X, float* __restrict__ sums,
                         float* __restrict__ sumsq, int M, int Nc)
{
  int c4 = threadIdx.x * 4;            // blockDim.x == Nc/4
  long r0 = (long)blockIdx.x * 64;
  float4 s = {0.f, 0.f, 0.f, 0.f}, q = {0.f, 0.f, 0.f, 0.f};
  for (int i = 0; i < 64; ++i) {
    long r = r0 + i;
    if (r < M) {
      float4 x = *(const float4*)(X + r * Nc + c4);
      s.x += x.x; s.y += x.y; s.z += x.z; s.w += x.w;
      q.x += x.x * x.x; q.y += x.y * x.y; q.z += x.z * x.z; q.w += x.w * x.w;
    }
  }
  atomicAdd(&sums[c4 + 0], s.x); atomicAdd(&sums[c4 + 1], s.y);
  atomicAdd(&sums[c4 + 2], s.z); atomicAdd(&sums[c4 + 3], s.w);
  atomicAdd(&sumsq[c4 + 0], q.x); atomicAdd(&sumsq[c4 + 1], q.y);
  atomicAdd(&sumsq[c4 + 2], q.z); atomicAdd(&sumsq[c4 + 3], q.w);
}

__global__ void bn_apply(float* __restrict__ X, const float* __restrict__ sums,
                         const float* __restrict__ sumsq,
                         const float* __restrict__ g, const float* __restrict__ b,
                         long M, int Nc, int relu)
{
  long t = (long)blockIdx.x * blockDim.x + threadIdx.x;
  long total = M * (long)(Nc >> 2);
  if (t >= total) return;
  int c4 = (int)(t % (Nc >> 2)) * 4;
  float invM = 1.f / (float)M;
  float4 sm = *(const float4*)(sums + c4);
  float4 sq = *(const float4*)(sumsq + c4);
  float4 gg = *(const float4*)(g + c4);
  float4 bb = *(const float4*)(b + c4);
  float4 x = ((const float4*)X)[t];
  float4 y;
  {
    float m = sm.x * invM, v = sq.x * invM - m * m;
    y.x = (x.x - m) * rsqrtf(v + 1e-5f) * gg.x + bb.x;
    m = sm.y * invM; v = sq.y * invM - m * m;
    y.y = (x.y - m) * rsqrtf(v + 1e-5f) * gg.y + bb.y;
    m = sm.z * invM; v = sq.z * invM - m * m;
    y.z = (x.z - m) * rsqrtf(v + 1e-5f) * gg.z + bb.z;
    m = sm.w * invM; v = sq.w * invM - m * m;
    y.w = (x.w - m) * rsqrtf(v + 1e-5f) * gg.w + bb.w;
  }
  if (relu) {
    y.x = fmaxf(y.x, 0.f); y.y = fmaxf(y.y, 0.f);
    y.z = fmaxf(y.z, 0.f); y.w = fmaxf(y.w, 0.f);
  }
  ((float4*)X)[t] = y;
}

// ---------------------------------------------------------------------------
// Irregular data movement (float4-vectorized; all col counts %4==0)
// ---------------------------------------------------------------------------
// out[didx[i]*ldo + ocol + c] += scale * in[sidx[i]*ldi + c]
__global__ void scatter_add_k(const float* __restrict__ in,
                              const int* __restrict__ sidx,
                              const int* __restrict__ didx,
                              float* __restrict__ out, long nmsg, int cols,
                              int ldi, int ldo, int ocol, float scale)
{
  int cols4 = cols >> 2;
  long t = (long)blockIdx.x * blockDim.x + threadIdx.x;
  if (t >= nmsg * (long)cols4) return;
  long i = t / cols4; int c = (int)(t % cols4) * 4;
  long sr = sidx ? (long)sidx[i] : i;
  long dr = didx ? (long)didx[i] : i;
  float4 v = *(const float4*)(in + sr * ldi + c);
  float* o = out + dr * ldo + ocol + c;
  atomicAdd(o + 0, scale * v.x);
  atomicAdd(o + 1, scale * v.y);
  atomicAdd(o + 2, scale * v.z);
  atomicAdd(o + 3, scale * v.w);
}

// out[r*ldo + ocol + c] (=|+=) in[idx[r]*ldi + c]
__global__ void gather_k(const float* __restrict__ in, const int* __restrict__ idx,
                         float* __restrict__ out, long rows, int cols,
                         int ldi, int ldo, int ocol, int addto)
{
  int cols4 = cols >> 2;
  long t = (long)blockIdx.x * blockDim.x + threadIdx.x;
  if (t >= rows * (long)cols4) return;
  long r = t / cols4; int c = (int)(t % cols4) * 4;
  long sr = idx ? (long)idx[r] : r;
  float4 v = *(const float4*)(in + sr * ldi + c);
  float4* o = (float4*)(out + r * ldo + ocol + c);
  if (addto) {
    float4 u = *o;
    v.x += u.x; v.y += u.y; v.z += u.z; v.w += u.w;
  }
  *o = v;
}

// out = (1 + *e) * x + y    (out may alias x or y); n%4==0
__global__ void axpy_e_k(float* __restrict__ out, const float* __restrict__ x,
                         const float* __restrict__ y, const float* __restrict__ e,
                         long n)
{
  long t = (long)blockIdx.x * blockDim.x + threadIdx.x;
  if (t >= (n >> 2)) return;
  float s = 1.f + *e;
  float4 xv = ((const float4*)x)[t];
  float4 yv = ((const float4*)y)[t];
  float4 o = {s * xv.x + yv.x, s * xv.y + yv.y, s * xv.z + yv.z, s * xv.w + yv.w};
  ((float4*)out)[t] = o;
}

// out = (1 + *e1)*x1 + (1 + *e2)*x2 + x3   (out may alias x3); n%4==0
__global__ void axpy3_e_k(float* __restrict__ out, const float* __restrict__ x1,
                          const float* __restrict__ e1, const float* __restrict__ x2,
                          const float* __restrict__ e2, const float* __restrict__ x3,
                          long n)
{
  long t = (long)blockIdx.x * blockDim.x + threadIdx.x;
  if (t >= (n >> 2)) return;
  float s1 = 1.f + *e1, s2 = 1.f + *e2;
  float4 a = ((const float4*)x1)[t];
  float4 b = ((const float4*)x2)[t];
  float4 d = ((const float4*)x3)[t];
  float4 o = {s1 * a.x + s2 * b.x + d.x, s1 * a.y + s2 * b.y + d.y,
              s1 * a.z + s2 * b.z + d.z, s1 * a.w + s2 * b.w + d.w};
  ((float4*)out)[t] = o;
}

// out[i] = a[b[i]]
__global__ void compose_idx_k(int* __restrict__ out, const int* __restrict__ a,
                              const int* __restrict__ b, long n)
{
  long t = (long)blockIdx.x * blockDim.x + threadIdx.x;
  if (t >= n) return;
  out[t] = a[b[t]];
}

// ---------------------------------------------------------------------------
// Host-side helpers
// ---------------------------------------------------------------------------
static inline unsigned gb(long total, int bs) {
  return (unsigned)((total + bs - 1) / bs);
}

static void gemm(const float* A, const float* W, float* C, long M, int K, int Nc,
                 int acc, _Float16* packbuf, hipStream_t s)
{
  pack_w<<<gb((long)K * Nc, 256), 256, 0, s>>>(W, packbuf, K, Nc);
  long tiles = (M / 16) * (long)(Nc / 64);
  gemm_wmma<<<dim3(gb(tiles, 8)), 256, 0, s>>>(A, packbuf, C, (int)M, K, Nc, acc);
}

static void bnrelu(float* X, const float* g, const float* b, long M, int Nc,
                   int relu, float* stats, hipStream_t s)
{
  hipMemsetAsync(stats, 0, 2 * Nc * sizeof(float), s);
  bn_stats<<<dim3(gb(M, 64)), Nc / 4, 0, s>>>(X, stats, stats + Nc, (int)M, Nc);
  long total = M * (long)(Nc / 4);
  bn_apply<<<dim3(gb(total, 256)), 256, 0, s>>>(X, stats, stats + Nc, g, b, M, Nc,
                                                relu);
}

extern "C" void kernel_launch(void* const* d_in, const int* in_sizes, int n_in,
                              void* d_out, int out_size, void* d_ws, size_t ws_size,
                              hipStream_t stream)
{
  (void)in_sizes; (void)n_in; (void)out_size; (void)ws_size;

  const int N = 100000, E = 200000, NC = 50000, RC = 300000;
  const int MNE = 400000, MEC = 300000, KEC = 600000, MCC = 200000, KCC = 400000;

  const float* node_rep  = (const float*)d_in[0];
  const float* edge_rep  = (const float*)d_in[1];
  const float* cycle_rep = (const float*)d_in[2];
  // params: jax pytree (dict) flattening = sorted keys (ASCII: 'W' < 'b' < 'g')
  const float* cc_W       = (const float*)d_in[3];
  const float* cc_b0      = (const float*)d_in[4];
  const float* cc_b1      = (const float*)d_in[5];
  const float* cc_g0      = (const float*)d_in[6];
  const float* cc_g1      = (const float*)d_in[7];
  const float* cc_lm_id   = (const float*)d_in[8];
  const float* cc_lm_inv  = (const float*)d_in[9];
  const float* cc_tf_int  = (const float*)d_in[10];
  const float* cc_tf_inv  = (const float*)d_in[11];
  const float* ec_e11     = (const float*)d_in[12];
  const float* ec_e12     = (const float*)d_in[13];
  const float* ec_e2      = (const float*)d_in[14];
  const float* ec_lift_W1 = (const float*)d_in[15];
  const float* ec_lift_W2 = (const float*)d_in[16];
  const float* ec_lift_b1 = (const float*)d_in[17];
  const float* ec_lift_b2 = (const float*)d_in[18];
  const float* ec_lift_g1 = (const float*)d_in[19];
  const float* ec_lift_g2 = (const float*)d_in[20];
  const float* ec_lvl1_W  = (const float*)d_in[21];
  const float* ec_lvl1_b  = (const float*)d_in[22];
  const float* ec_lvl1_g  = (const float*)d_in[23];
  const float* ec_lvl2_W1 = (const float*)d_in[24];
  const float* ec_lvl2_W2 = (const float*)d_in[25];
  const float* ec_lvl2_b1 = (const float*)d_in[26];
  const float* ec_lvl2_b2 = (const float*)d_in[27];
  const float* ec_lvl2_g1 = (const float*)d_in[28];
  const float* ec_lvl2_g2 = (const float*)d_in[29];
  const float* mlp_W      = (const float*)d_in[30];
  const float* mlp_b      = (const float*)d_in[31];
  const float* mlp_g      = (const float*)d_in[32];
  const float* mlpc_W     = (const float*)d_in[33];
  const float* mlpc_b     = (const float*)d_in[34];
  const float* mlpc_g     = (const float*)d_in[35];
  const float* ne_e1      = (const float*)d_in[36];
  const float* ne_e2      = (const float*)d_in[37];
  const float* ne_lift_W1 = (const float*)d_in[38];
  const float* ne_lift_W2 = (const float*)d_in[39];
  const float* ne_lift_b1 = (const float*)d_in[40];
  const float* ne_lift_b2 = (const float*)d_in[41];
  const float* ne_lift_g1 = (const float*)d_in[42];
  const float* ne_lift_g2 = (const float*)d_in[43];
  const float* ne_lvl1_W  = (const float*)d_in[44];
  const float* ne_lvl1_b  = (const float*)d_in[45];
  const float* ne_lvl1_g  = (const float*)d_in[46];
  const float* ne_lvl2_W1 = (const float*)d_in[47];
  const float* ne_lvl2_W2 = (const float*)d_in[48];
  const float* ne_lvl2_b1 = (const float*)d_in[49];
  const float* ne_lvl2_b2 = (const float*)d_in[50];
  const float* ne_lvl2_g1 = (const float*)d_in[51];
  const float* ne_lvl2_g2 = (const float*)d_in[52];
  const int* ne_src     = (const int*)d_in[53];
  const int* ne_dst     = (const int*)d_in[54];
  const int* cyc_dom    = (const int*)d_in[55];
  const int* ec_msg_src = (const int*)d_in[56];
  const int* ec_msg_tgt = (const int*)d_in[57];
  const int* ec_int_msg = (const int*)d_in[58];
  const int* ec_int_row = (const int*)d_in[59];
  const int* cc_msg_src = (const int*)d_in[60];
  const int* cc_msg_tgt = (const int*)d_in[61];
  const int* cc_int_msg = (const int*)d_in[62];
  const int* cc_src_row = (const int*)d_in[63];
  const int* cc_tgt_row = (const int*)d_in[64];

  // ---- workspace (bump allocator, 128B-aligned chunks) ----
  float* ws = (float*)d_ws;
  size_t off = 0;
  auto WA = [&](size_t elems) {
    float* p = ws + off;
    off += (elems + 31) & ~((size_t)31);
    return p;
  };
  float* stats    = WA(512);
  _Float16* w_pack = (_Float16*)WA(49152);   // max K*Nc halfs = 384*256
  int*   fusedEC  = (int*)WA(KEC);
  int*   fusedCC  = (int*)WA(KCC);
  float* w_lift00 = WA((size_t)E * HH);    // lift00 -> (1+e2)*edge+lift00
  float* w_catA   = WA((size_t)E * HH2);   // [E,2H] concat / hidden scratch
  float* w_lvlE   = WA((size_t)E * HH);    // lvl_e -> edge2
  float* w_lvlN   = WA((size_t)N * HH);    // lvl_n/node_in -> y_inv [NC,2H]
  float* w_h2N    = WA((size_t)N * HH2);   // node hidden -> inv_maps [NC,2H]
  float* w_edge1  = WA((size_t)E * HH);
  float* w_sumNC  = WA((size_t)NC * HH);
  float* w_lift01 = WA((size_t)RC * HH2);  // lift01 -> cyc hidden -> cyc_cat
  float* w_catC   = WA((size_t)RC * HH3);  // [lift01|cycle] -> y_int
  float* w_lvlC   = WA((size_t)RC * HH);   // lvl_c -> cc2
  float* w_a      = WA((size_t)MEC * HH);  // a -> s_int
  float* w_bmean  = WA((size_t)NC * HH);
  float* w_r0     = WA((size_t)E * HH);
  float* w_r1     = WA((size_t)E * HH);    // -> edge2 input
  float* w_cmean  = WA((size_t)NC * HH);   // == x_dom
  float* w_lm11   = WA((size_t)RC * HH2);  // lm11/cyc_in -> irred/cc0/cc1
  float* w_csplit = WA((size_t)RC * HH);

  float* node_out = (float*)d_out;
  float* edge_out = (float*)d_out + (size_t)N * HH;
  float* cyc_out  = (float*)d_out + (size_t)(N + E) * HH;

  // ================= Phase A: nodes <-> edges =================
  hipMemsetAsync(w_lift00, 0, (size_t)E * HH * 4, stream);
  scatter_add_k<<<gb((long)MNE * 32, 256), 256, 0, stream>>>(
      node_rep, ne_src, ne_dst, w_lift00, MNE, HH, HH, HH, 0, 1.f);
  gather_k<<<gb((long)E * 32, 256), 256, 0, stream>>>(
      w_lift00, nullptr, w_catA, E, HH, HH, HH2, 0, 0);
  gather_k<<<gb((long)E * 32, 256), 256, 0, stream>>>(
      edge_rep, nullptr, w_catA, E, HH, HH, HH2, HH, 0);
  gemm(w_catA, ne_lvl1_W, w_lvlE, E, HH2, HH, 0, w_pack, stream);
  bnrelu(w_lvlE, ne_lvl1_g, ne_lvl1_b, E, HH, 1, stats, stream);
  hipMemsetAsync(w_lvlN, 0, (size_t)N * HH * 4, stream);
  scatter_add_k<<<gb((long)MNE * 32, 256), 256, 0, stream>>>(
      w_lvlE, ne_dst, ne_src, w_lvlN, MNE, HH, HH, HH, 0, 1.f);
  axpy_e_k<<<gb((long)N * 32, 256), 256, 0, stream>>>(
      w_lvlN, node_rep, w_lvlN, ne_e1, (long)N * HH);
  // node_out = mlp2(node_in, ne_lvl2)
  gemm(w_lvlN, ne_lvl2_W1, w_h2N, N, HH, HH2, 0, w_pack, stream);
  bnrelu(w_h2N, ne_lvl2_g1, ne_lvl2_b1, N, HH2, 1, stats, stream);
  gemm(w_h2N, ne_lvl2_W2, node_out, N, HH2, HH, 0, w_pack, stream);
  bnrelu(node_out, ne_lvl2_g2, ne_lvl2_b2, N, HH, 1, stats, stream);
  // edge1 = mlp2((1+ne_e2)*edge_rep + lift00, ne_lift)
  axpy_e_k<<<gb((long)E * 32, 256), 256, 0, stream>>>(
      w_lift00, edge_rep, w_lift00, ne_e2, (long)E * HH);
  gemm(w_lift00, ne_lift_W1, w_catA, E, HH, HH2, 0, w_pack, stream);
  bnrelu(w_catA, ne_lift_g1, ne_lift_b1, E, HH2, 1, stats, stream);
  gemm(w_catA, ne_lift_W2, w_edge1, E, HH2, HH, 0, w_pack, stream);
  bnrelu(w_edge1, ne_lift_g2, ne_lift_b2, E, HH, 1, stats, stream);

  // ================= Phase B: edges <-> cycles =================
  hipMemsetAsync(w_sumNC, 0, (size_t)NC * HH * 4, stream);
  scatter_add_k<<<gb((long)MEC * 32, 256), 256, 0, stream>>>(
      edge_rep, ec_msg_src, ec_msg_tgt, w_sumNC, MEC, HH, HH, HH, 0, 1.f);
  compose_idx_k<<<gb(KEC, 256), 256, 0, stream>>>(fusedEC, ec_msg_src, ec_int_msg,
                                                  KEC);
  hipMemsetAsync(w_lift01, 0, (size_t)RC * HH2 * 4, stream);
  scatter_add_k<<<gb((long)KEC * 32, 256), 256, 0, stream>>>(
      edge_rep, fusedEC, ec_int_row, w_lift01, KEC, HH, HH, HH2, 0, 1.f);
  gather_k<<<gb((long)RC * 32, 256), 256, 0, stream>>>(
      w_sumNC, cyc_dom, w_lift01, RC, HH, HH, HH2, HH, 0);
  gather_k<<<gb((long)RC * 64, 256), 256, 0, stream>>>(
      w_lift01, nullptr, w_catC, RC, HH2, HH2, HH3, 0, 0);
  gather_k<<<gb((long)RC * 32, 256), 256, 0, stream>>>(
      cycle_rep, nullptr, w_catC, RC, HH, HH, HH3, HH2, 0);
  gemm(w_catC, ec_lvl1_W, w_lvlC, RC, HH3, HH, 0, w_pack, stream);
  bnrelu(w_lvlC, ec_lvl1_g, ec_lvl1_b, RC, HH, 1, stats, stream);
  // a = seg_mean(lvl_c[ec_int_row], ec_int_msg)  (count==2 -> scale 0.5)
  hipMemsetAsync(w_a, 0, (size_t)MEC * HH * 4, stream);
  scatter_add_k<<<gb((long)KEC * 32, 256), 256, 0, stream>>>(
      w_lvlC, ec_int_row, ec_int_msg, w_a, KEC, HH, HH, HH, 0, 0.5f);
  // bmean = seg_mean(lvl_c, cyc_dom) (count==6)
  hipMemsetAsync(w_bmean, 0, (size_t)NC * HH * 4, stream);
  scatter_add_k<<<gb((long)RC * 32, 256), 256, 0, stream>>>(
      w_lvlC, nullptr, cyc_dom, w_bmean, RC, HH, HH, HH, 0, 1.f / 6.f);
  hipMemsetAsync(w_r0, 0, (size_t)E * HH * 4, stream);
  scatter_add_k<<<gb((long)MEC * 32, 256), 256, 0, stream>>>(
      w_a, nullptr, ec_msg_src, w_r0, MEC, HH, HH, HH, 0, 1.f);
  hipMemsetAsync(w_r1, 0, (size_t)E * HH * 4, stream);
  scatter_add_k<<<gb((long)MEC * 32, 256), 256, 0, stream>>>(
      w_bmean, ec_msg_tgt, ec_msg_src, w_r1, MEC, HH, HH, HH, 0, 1.f);
  axpy3_e_k<<<gb((long)E * 32, 256), 256, 0, stream>>>(
      w_r1, edge_rep, ec_e11, w_r0, ec_e12, w_r1, (long)E * HH);
  // edge2 = mlp2(w_r1, ec_lvl2)
  gemm(w_r1, ec_lvl2_W1, w_catA, E, HH, HH2, 0, w_pack, stream);
  bnrelu(w_catA, ec_lvl2_g1, ec_lvl2_b1, E, HH2, 1, stats, stream);
  gemm(w_catA, ec_lvl2_W2, w_lvlE, E, HH2, HH, 0, w_pack, stream);
  bnrelu(w_lvlE, ec_lvl2_g2, ec_lvl2_b2, E, HH, 1, stats, stream);
  // cmean == x_dom = seg_mean(cycle_rep, cyc_dom)
  hipMemsetAsync(w_cmean, 0, (size_t)NC * HH * 4, stream);
  scatter_add_k<<<gb((long)RC * 32, 256), 256, 0, stream>>>(
      cycle_rep, nullptr, cyc_dom, w_cmean, RC, HH, HH, HH, 0, 1.f / 6.f);
  // lm11 = [cycle_rep | cmean[cyc_dom]] ; cyc_in = (1+ec_e2)*lm11 + lift01
  gather_k<<<gb((long)RC * 32, 256), 256, 0, stream>>>(
      cycle_rep, nullptr, w_lm11, RC, HH, HH, HH2, 0, 0);
  gather_k<<<gb((long)RC * 32, 256), 256, 0, stream>>>(
      w_cmean, cyc_dom, w_lm11, RC, HH, HH, HH2, HH, 0);
  axpy_e_k<<<gb((long)RC * 64, 256), 256, 0, stream>>>(
      w_lm11, w_lm11, w_lift01, ec_e2, (long)RC * HH2);
  // cyc_split = mlp2(cyc_in, ec_lift) — hidden reuses w_lift01
  gemm(w_lm11, ec_lift_W1, w_lift01, RC, HH2, HH2, 0, w_pack, stream);
  bnrelu(w_lift01, ec_lift_g1, ec_lift_b1, RC, HH2, 1, stats, stream);
  gemm(w_lift01, ec_lift_W2, w_csplit, RC, HH2, HH, 0, w_pack, stream);
  bnrelu(w_csplit, ec_lift_g2, ec_lift_b2, RC, HH, 1, stats, stream);
  // edge_out = lbr([edge1|edge2] @ mlp_W)
  gather_k<<<gb((long)E * 32, 256), 256, 0, stream>>>(
      w_edge1, nullptr, w_catA, E, HH, HH, HH2, 0, 0);
  gather_k<<<gb((long)E * 32, 256), 256, 0, stream>>>(
      w_lvlE, nullptr, w_catA, E, HH, HH, HH2, HH, 0);
  gemm(w_catA, mlp_W, edge_out, E, HH2, HH, 0, w_pack, stream);
  bnrelu(edge_out, mlp_g, mlp_b, E, HH, 1, stats, stream);

  // ================= Phase C: cycles <-> cycles =================
  float* w_sint = w_a;        // [MCC,H] fits in [MEC,H]
  hipMemsetAsync(w_sint, 0, (size_t)MCC * HH * 4, stream);
  scatter_add_k<<<gb((long)KCC * 32, 256), 256, 0, stream>>>(
      cycle_rep, cc_src_row, cc_int_msg, w_sint, KCC, HH, HH, HH, 0, 1.f);
  float* w_yint = w_catC;     // [RC,3H]
  hipMemsetAsync(w_yint, 0, (size_t)RC * HH3 * 4, stream);
  scatter_add_k<<<gb((long)KCC * 32, 256), 256, 0, stream>>>(
      cycle_rep, cc_src_row, cc_tgt_row, w_yint, KCC, HH, HH, HH3, 0, 1.f);
  scatter_add_k<<<gb((long)KCC * 32, 256), 256, 0, stream>>>(
      w_sint, cc_int_msg, cc_tgt_row, w_yint, KCC, HH, HH, HH3, HH, 1.f);
  compose_idx_k<<<gb(KCC, 256), 256, 0, stream>>>(fusedCC, cc_msg_src, cc_int_msg,
                                                  KCC);
  scatter_add_k<<<gb((long)KCC * 32, 256), 256, 0, stream>>>(
      w_cmean, fusedCC, cc_tgt_row, w_yint, KCC, HH, HH, HH3, HH2, 1.f);
  float* w_yinv = w_lvlN;     // [NC,2H] == N*H floats
  hipMemsetAsync(w_yinv, 0, (size_t)NC * HH2 * 4, stream);
  scatter_add_k<<<gb((long)MCC * 32, 256), 256, 0, stream>>>(
      w_sint, nullptr, cc_msg_tgt, w_yinv, MCC, HH, HH, HH2, 0, 1.f);
  scatter_add_k<<<gb((long)MCC * 32, 256), 256, 0, stream>>>(
      w_cmean, cc_msg_src, cc_msg_tgt, w_yinv, MCC, HH, HH, HH2, HH, 1.f);
  // inv_maps = x_dom @ cc_lm_inv + y_inv @ cc_tf_inv
  float* w_invm = w_h2N;
  gemm(w_cmean, cc_lm_inv, w_invm, NC, HH, HH2, 0, w_pack, stream);
  gemm(w_yinv, cc_tf_inv, w_invm, NC, HH2, HH2, 1, w_pack, stream);
  // irred = cycle_rep @ cc_lm_id + y_int @ cc_tf_int
  float* w_irred = w_lm11;
  gemm(cycle_rep, cc_lm_id, w_irred, RC, HH, HH2, 0, w_pack, stream);
  gemm(w_yint, cc_tf_int, w_irred, RC, HH3, HH2, 1, w_pack, stream);
  // cc0 = irred + inv_maps[cyc_dom]; cc1 = relu(bn(cc0))
  gather_k<<<gb((long)RC * 64, 256), 256, 0, stream>>>(
      w_invm, cyc_dom, w_irred, RC, HH2, HH2, HH2, 0, 1);
  bnrelu(w_irred, cc_g0, cc_b0, RC, HH2, 1, stats, stream);
  // cc2 = lbr(cc1 @ cc_W)
  float* w_cc2 = w_lvlC;
  gemm(w_irred, cc_W, w_cc2, RC, HH2, HH, 0, w_pack, stream);
  bnrelu(w_cc2, cc_g1, cc_b1, RC, HH, 1, stats, stream);
  // cycle_out = lbr([cyc_split|cc2] @ mlpc_W)
  float* w_ccat = w_lift01;
  gather_k<<<gb((long)RC * 32, 256), 256, 0, stream>>>(
      w_csplit, nullptr, w_ccat, RC, HH, HH, HH2, 0, 0);
  gather_k<<<gb((long)RC * 32, 256), 256, 0, stream>>>(
      w_cc2, nullptr, w_ccat, RC, HH, HH, HH2, HH, 0);
  gemm(w_ccat, mlpc_W, cyc_out, RC, HH2, HH, 0, w_pack, stream);
  bnrelu(cyc_out, mlpc_g, mlpc_b, RC, HH, 1, stats, stream);
}
